// RangeLoss_62027917689087
// MI455X (gfx1250) — matrix operand order repaired and checked by reading, not compiled, not run on hardware
//
#include <hip/hip_runtime.h>
#include <hip/hip_bf16.h>

typedef __attribute__((ext_vector_type(16))) _Float16 v16h;
typedef __attribute__((ext_vector_type(8)))  _Float16 v8h;
typedef __attribute__((ext_vector_type(4)))  _Float16 v4h;
typedef __attribute__((ext_vector_type(8)))  float    v8f;

#define CLASSES 64
#define ROWS    1024          // samples per class (BATCH / NUM_CLASSES)
#define DIM     64
#define TILE    16
#define NBLK    (ROWS / TILE) // 64 block rows/cols
#define EPSF    1e-12f
#define NEGBIG  (-1.0e30f)

// branchless top-2 tracker: after this, t1 >= t2 >= all other seen values
__device__ __forceinline__ void top2_update(float v, float& t1, float& t2) {
    t2 = fmaxf(t2, fminf(t1, v));
    t1 = fmaxf(t1, v);
}

// ---------------------------------------------------------------------------
// Kernel 1: one workgroup per class.
//  Phase A: stage class features f32->f16 into LDS, per-row |x|^2 (f32), center.
//  Phase B: upper-triangular 16x16 Gram tiles via v_wmma_f32_16x16x32_f16,
//           fused d^2 = sq_i + sq_j - 2*G with branchless per-thread top-2.
//  Phase C: deterministic block top-2 reduction -> harmonic intra loss per class.
// ---------------------------------------------------------------------------
extern "C" __global__ __launch_bounds__(256) void range_intra_kernel(
    const float* __restrict__ feats,
    float* __restrict__ intra_out,      // [CLASSES]
    float* __restrict__ centers_out)    // [CLASSES * DIM]
{
    extern __shared__ char smem_raw[];
    _Float16* Xh    = (_Float16*)smem_raw;                   // 1024*64 f16 = 128 KB
    float*    sq_s  = (float*)(smem_raw + ROWS * DIM * 2);   // 1024 f32   = 4 KB
    float*    red   = sq_s + ROWS;                           // 512 f32    = 2 KB
    float*    cpart = red + 512;                             // 256 f32    = 1 KB

    const int c = blockIdx.x;
    const int t = threadIdx.x;
    const float* X = feats + ((size_t)c * ROWS) * DIM;

    // -------- Phase A: stage + sq --------
    for (int rr = 0; rr < 4; ++rr) {
        int row = t * 4 + rr;
        const float4* src = (const float4*)(X + row * DIM);
        float s = 0.f;
        for (int q = 0; q < DIM / 4; ++q) {
            float4 v = src[q];
            s += v.x * v.x + v.y * v.y + v.z * v.z + v.w * v.w;
            v4h h = { (_Float16)v.x, (_Float16)v.y, (_Float16)v.z, (_Float16)v.w };
            *(v4h*)(Xh + row * DIM + q * 4) = h;     // 8B aligned -> ds_store_b64
        }
        sq_s[row] = s;
    }
    __syncthreads();

    // -------- class center (f16 sums fine: inter loss is relu around 0.1) ----
    {
        int f = t & 63, blk = t >> 6;
        float acc = 0.f;
        for (int r = blk * 256; r < blk * 256 + 256; ++r)
            acc += (float)Xh[r * DIM + f];
        cpart[t] = acc;
    }
    __syncthreads();
    if (t < 64) {
        float s = cpart[t] + cpart[t + 64] + cpart[t + 128] + cpart[t + 192];
        centers_out[c * DIM + t] = s * (1.0f / (float)ROWS);
    }

    // -------- Phase B: WMMA Gram tiles, upper triangle only --------
    const int wave = t >> 5;
    const int lane = t & 31;
    const int lhi  = lane >> 4;   // 0 or 1 (half-wave)
    const int llo  = lane & 15;

    float top1 = NEGBIG, top2 = NEGBIG;

    // diagonal-tile dedup mask is bi-independent: element (r) is valid iff
    // col llo > row (lhi*8 + r). Precompute as 8 lane-constant floats.
    float diagsel[8];
    for (int r = 0; r < 8; ++r)
        diagsel[r] = (llo > lhi * 8 + r) ? 0.0f : NEGBIG;  // add-mask trick unused; select below

    for (int round = 0; round < NBLK / 8; ++round) {
        // reverse wave->row mapping on odd rounds: 260 tiles per wave, balanced
        int w  = (round & 1) ? (7 - wave) : wave;
        int bi = round * 8 + w;

        // A fragment (rows bi*16..+15, K = 0..63), contiguous-k 16x32 layout (B matches):
        // lane llo(+16*lhi): row llo, elems [ab..ab+15], ab = 16*lhi
        const _Float16* arow = Xh + (bi * TILE + llo) * DIM;
        const int ab = lhi * 16;
        v16h a0 = {}, a1 = {};
        {
            v8h lo0 = *(const v8h*)(arow + ab);
            v8h hi0 = *(const v8h*)(arow + ab + 8);
            v8h lo1 = *(const v8h*)(arow + ab + 32);
            v8h hi1 = *(const v8h*)(arow + ab + 40);
            for (int i = 0; i < 8; ++i) {
                a0[i] = lo0[i]; a0[8 + i] = hi0[i];
                a1[i] = lo1[i]; a1[8 + i] = hi1[i];
            }
        }

        // sq_i for this block-row: 8 contiguous f32 -> two ds_load_b128
        float sqi[8];
        {
            float4 s0 = *(const float4*)(sq_s + bi * TILE + lhi * 8);
            float4 s1 = *(const float4*)(sq_s + bi * TILE + lhi * 8 + 4);
            sqi[0] = s0.x; sqi[1] = s0.y; sqi[2] = s0.z; sqi[3] = s0.w;
            sqi[4] = s1.x; sqi[5] = s1.y; sqi[6] = s1.z; sqi[7] = s1.w;
        }

        // ---- diagonal tile (bj == bi): masked, wave-uniform special case ----
        {
            const _Float16* brow = arow;           // same block of rows
            const int kb = lhi * 16;
            v16h b0 = *(const v16h*)(brow + kb);
            v16h b1 = *(const v16h*)(brow + kb + 32);
            v8f acc = {};
            acc = __builtin_amdgcn_wmma_f32_16x16x32_f16(
                      false, a0, false, b0, (short)0, acc, false, false);
            acc = __builtin_amdgcn_wmma_f32_16x16x32_f16(
                      false, a1, false, b1, (short)0, acc, false, false);
            float sqj = sq_s[bi * TILE + llo];
            for (int r = 0; r < 8; ++r) {
                float d2 = __builtin_fmaf(-2.0f, acc[r], sqi[r] + sqj);
                d2 = (diagsel[r] == 0.0f) ? d2 : NEGBIG;   // v_cndmask, no branch
                top2_update(d2, top1, top2);
            }
        }

        // ---- strictly-upper tiles: unmasked, branch-free inner body ----
        for (int bj = bi + 1; bj < NBLK; ++bj) {
            const _Float16* brow = Xh + (bj * TILE + llo) * DIM;
            const int kb = lhi * 16;
            v16h b0 = *(const v16h*)(brow + kb);
            v16h b1 = *(const v16h*)(brow + kb + 32);
            v8f acc = {};
            acc = __builtin_amdgcn_wmma_f32_16x16x32_f16(
                      false, a0, false, b0, (short)0, acc, false, false);
            acc = __builtin_amdgcn_wmma_f32_16x16x32_f16(
                      false, a1, false, b1, (short)0, acc, false, false);
            float sqj = sq_s[bj * TILE + llo];
            for (int r = 0; r < 8; ++r) {
                float d2 = __builtin_fmaf(-2.0f, acc[r], sqi[r] + sqj);
                top2_update(d2, top1, top2);
            }
        }
    }

    // -------- Phase C: deterministic block top-2 reduce --------
    red[t * 2 + 0] = top1;
    red[t * 2 + 1] = top2;
    __syncthreads();
    if (t == 0) {
        float t1 = NEGBIG, t2 = NEGBIG;
        for (int i = 0; i < 512; ++i)
            top2_update(red[i], t1, t2);
        float d1 = __builtin_sqrtf(fmaxf(t1, EPSF));
        float d2 = __builtin_sqrtf(fmaxf(t2, EPSF));
        // K / sum(1/topk), K = 2
        intra_out[c] = 2.0f / (1.0f / d1 + 1.0f / d2);
    }
}

// ---------------------------------------------------------------------------
// Kernel 2: single workgroup. Min inter-center distance + final scalar.
// sorted(cdist)[C] == min off-diagonal (the C diagonal entries are 1e-6).
// ---------------------------------------------------------------------------
extern "C" __global__ __launch_bounds__(256) void range_final_kernel(
    const float* __restrict__ centers,
    const float* __restrict__ intra,
    float* __restrict__ out)
{
    __shared__ float Cs[64 * 64];
    __shared__ float csq[64];
    __shared__ float mins[256];
    const int t = threadIdx.x;

    for (int i = t; i < 64 * 64; i += 256) Cs[i] = centers[i];
    __syncthreads();
    if (t < 64) {
        float s = 0.f;
        for (int k = 0; k < 64; ++k) { float v = Cs[t * 64 + k]; s += v * v; }
        csq[t] = s;
    }
    __syncthreads();

    float mn = 3.4e38f;
    for (int p = t; p < 64 * 64; p += 256) {
        int i = p >> 6, j = p & 63;
        float dot = 0.f;
        for (int k = 0; k < 64; ++k) dot += Cs[i * 64 + k] * Cs[j * 64 + k];
        float d2 = csq[i] + csq[j] - 2.0f * dot;
        float d  = __builtin_sqrtf(fmaxf(d2, EPSF));
        mn = fminf(mn, (i == j) ? 3.4e38f : d);   // branchless diagonal skip
    }
    mins[t] = mn;
    __syncthreads();
    if (t == 0) {
        float m = mins[0];
        for (int i = 1; i < 256; ++i) m = fminf(m, mins[i]);
        float inter = fmaxf(0.f, 0.1f - m);
        float s = 0.f;
        for (int c = 0; c < CLASSES; ++c) s += intra[c];   // fixed order: deterministic
        out[0] = 0.5f * s + 0.5f * inter;
    }
}

// ---------------------------------------------------------------------------
extern "C" void kernel_launch(void* const* d_in, const int* in_sizes, int n_in,
                              void* d_out, int out_size, void* d_ws, size_t ws_size,
                              hipStream_t stream) {
    (void)in_sizes; (void)n_in; (void)out_size; (void)ws_size;
    const float* feats = (const float*)d_in[0];   // [65536 * 64] f32
    // d_in[1] (targets) and d_in[2] (num_classes) are unused by the reference math
    float* centers = (float*)d_ws;                // 64*64 f32
    float* intra   = centers + CLASSES * DIM;     // 64 f32
    float* out     = (float*)d_out;

    size_t shmem = (size_t)ROWS * DIM * sizeof(_Float16)  // f16 features
                 + ROWS * sizeof(float)                   // sq
                 + 512  * sizeof(float)                   // top-2 reduction
                 + 256  * sizeof(float);                  // center partials

    range_intra_kernel<<<dim3(CLASSES), dim3(256), shmem, stream>>>(feats, intra, centers);
    range_final_kernel<<<dim3(1), dim3(256), 0, stream>>>(centers, intra, out);
}
